// LSTMSeq2Seq_75900662055087
// MI455X (gfx1250) — matrix operand order repaired and checked by reading, not compiled, not run on hardware
//
#include <hip/hip_runtime.h>

typedef __attribute__((ext_vector_type(16))) __bf16  v16bf;
typedef __attribute__((ext_vector_type(8)))  float   v8f;

#define DEVINL __device__ __forceinline__

constexpr int  BATCH   = 256;
constexpr int  TSEQ    = 256;
constexpr int  HID     = 512;
constexpr int  INF     = 32;
constexpr int  OUTF    = 8;
constexpr int  FUT     = 96;
constexpr int  GATES   = 4 * HID;     // 2048
constexpr int  NBLK    = 32;          // persistent grid (2 m-blocks x 16 n-blocks)
constexpr int  THREADS = 256;         // 8 wave32

// ---------------- workspace layout ----------------
constexpr size_t SZ_H  = (size_t)BATCH * HID * 2;   // bf16 hidden state
constexpr size_t SZ_C  = (size_t)BATCH * HID * 4;   // f32 cell state
constexpr size_t SZ_BC = (size_t)GATES * 4;         // combined bias
constexpr size_t SZ_PS = (size_t)GATES * 32  * 2;   // packed K<=32 weights (padded)
constexpr size_t SZ_PB = (size_t)GATES * HID * 2;   // packed K=512 weights

constexpr size_t OFF_SYNC   = 0;
constexpr size_t OFF_H0A    = 256;                  // H0A,H0B contiguous (ping-pong by offset)
constexpr size_t OFF_H1A    = OFF_H0A + 2 * SZ_H;   // H1A,H1B contiguous
constexpr size_t OFF_C0     = OFF_H1A + 2 * SZ_H;
constexpr size_t OFF_C1     = OFF_C0 + SZ_C;
constexpr size_t OFF_XD     = OFF_C1 + SZ_C;        // decoder autoregressive input (bf16 256x8)
constexpr size_t OFF_BC0    = OFF_XD + 8192;
constexpr size_t OFF_BC1    = OFF_BC0 + SZ_BC;
constexpr size_t OFF_BC2    = OFF_BC1 + SZ_BC;
constexpr size_t OFF_BC3    = OFF_BC2 + SZ_BC;
constexpr size_t OFF_PEWIH0 = OFF_BC3 + SZ_BC;
constexpr size_t OFF_PEWHH0 = OFF_PEWIH0 + SZ_PS;
constexpr size_t OFF_PEWIH1 = OFF_PEWHH0 + SZ_PB;
constexpr size_t OFF_PEWHH1 = OFF_PEWIH1 + SZ_PB;
constexpr size_t OFF_PDWIH0 = OFF_PEWHH1 + SZ_PB;
constexpr size_t OFF_PDWHH0 = OFF_PDWIH0 + SZ_PS;
constexpr size_t OFF_PDWIH1 = OFF_PDWHH0 + SZ_PB;
constexpr size_t OFF_PDWHH1 = OFF_PDWIH1 + SZ_PB;

// ---------------- helpers ----------------
DEVINL unsigned short f2bf(float f) {
  unsigned u = __float_as_uint(f);
  unsigned r = u + 0x7FFFu + ((u >> 16) & 1u);   // round-to-nearest-even
  return (unsigned short)(r >> 16);
}
DEVINL float bf2f(unsigned short h) { return __uint_as_float((unsigned)h << 16); }

DEVINL float sigf(float x) { return 1.f / (1.f + __expf(-x)); }
DEVINL float tanh_s(float x) {
  float ax = fabsf(x);
  float e  = __expf(-2.f * ax);
  float t  = (1.f - e) / (1.f + e);
  return copysignf(t, x);
}

union BF16Frag {
  v16bf v;
  uint4 q[2];
  unsigned short u[16];
};

DEVINL v8f wmma_bf16(const v16bf& a, const v16bf& b, const v8f& c) {
  return __builtin_amdgcn_wmma_f32_16x16x32_bf16(false, a, false, b, (short)0, c, false, false);
}

// packed B-fragment loader: weights pre-laid-out so each lane's 16 bf16 are contiguous
DEVINL void load_bfrag(BF16Frag b[4], const unsigned short* __restrict__ Wnb,
                       int kc, int wave_n, int lane) {
#pragma unroll
  for (int wn = 0; wn < 4; ++wn) {
    const unsigned short* bp = Wnb + ((size_t)(kc * 8 + wave_n * 4 + wn) * 32 + lane) * 16;
    b[wn].q[0] = *(const uint4*)(bp);
    b[wn].q[1] = *(const uint4*)(bp + 8);
  }
}

DEVINL void load_afrag(BF16Frag& a0, BF16Frag& a1, const unsigned short* __restrict__ ap) {
  a0.q[0] = *(const uint4*)(ap);
  a0.q[1] = *(const uint4*)(ap + 16);
  a1.q[0] = *(const uint4*)(ap + 16 * HID);
  a1.q[1] = *(const uint4*)(ap + 16 * HID + 16);
}

// K=512 bf16 GEMM accumulation with explicit register double-buffering:
// chunk kc+1's A/B fragments are in flight while chunk kc's 8 WMMAs issue.
DEVINL void gemm_h512(v8f acc[2][4], const unsigned short* __restrict__ A,
                      const unsigned short* __restrict__ Wp,
                      int rA, int kh, int wave_n, int n_blk, int lane) {
  const unsigned short* Wnb = Wp + (size_t)n_blk * (16 * 8 * 32 * 16);
  const unsigned short* ap  = A + (size_t)rA * HID + kh;

  BF16Frag a0[2], a1[2], b[2][4];
  load_afrag(a0[0], a1[0], ap);
  load_bfrag(b[0], Wnb, 0, wave_n, lane);

#pragma unroll
  for (int kc = 0; kc < 16; ++kc) {
    const int cur = kc & 1;
    const int nxt = cur ^ 1;
    if (kc < 15) {
      load_afrag(a0[nxt], a1[nxt], ap + (kc + 1) * 32);
      load_bfrag(b[nxt], Wnb, kc + 1, wave_n, lane);
    }
#pragma unroll
    for (int wn = 0; wn < 4; ++wn) {
      acc[0][wn] = wmma_bf16(a0[cur].v, b[cur][wn].v, acc[0][wn]);
      acc[1][wn] = wmma_bf16(a1[cur].v, b[cur][wn].v, acc[1][wn]);
    }
  }
}

// gate exchange through LDS + pointwise LSTM cell update
DEVINL void lstm_epilogue(v8f acc[2][4], const float* __restrict__ bc,
                          float* __restrict__ c, unsigned short* __restrict__ hout,
                          float* Gs, int m_base, int n_blk, int wave_m, int wave_n, int lane) {
  const int hi = lane >> 4;
  const int nl = lane & 15;
#pragma unroll
  for (int wm = 0; wm < 2; ++wm) {
#pragma unroll
    for (int wn = 0; wn < 4; ++wn) {
      int row0   = wave_m * 32 + wm * 16 + hi * 8;
      int col    = wave_n * 64 + wn * 16 + nl;
      float bias = bc[(col >> 5) * HID + n_blk * 32 + (col & 31)];
#pragma unroll
      for (int e = 0; e < 8; ++e)
        Gs[(row0 + e) * 132 + col] = acc[wm][wn][e] + bias;
    }
  }
  __syncthreads();
  for (int idx = threadIdx.x; idx < 128 * 32; idx += THREADS) {
    int r = idx >> 5, j = idx & 31;
    float iv = Gs[r * 132 +      j];
    float fv = Gs[r * 132 + 32 + j];
    float gv = Gs[r * 132 + 64 + j];
    float ov = Gs[r * 132 + 96 + j];
    size_t off = (size_t)(m_base + r) * HID + n_blk * 32 + j;
    float cold = c[off];
    float cn = sigf(fv) * cold + sigf(iv) * tanh_s(gv);
    float hn = sigf(ov) * tanh_s(cn);
    c[off]    = cn;
    hout[off] = f2bf(hn);
  }
}

// one LSTM layer step. XM: 0 = f32 x (K=32, encoder L0), 1 = bf16 x (K=512), 2 = bf16 x (K=8, decoder L0)
template <int XM>
DEVINL void lstm_step(const float* __restrict__ xf, int xstride,
                      const unsigned short* __restrict__ xh,
                      const unsigned short* __restrict__ Wx,
                      const unsigned short* __restrict__ Wh,
                      const float* __restrict__ bc,
                      const unsigned short* __restrict__ hp,
                      float* __restrict__ c, unsigned short* __restrict__ hout,
                      float* Gs) {
  const int lane   = threadIdx.x & 31;
  const int wid    = threadIdx.x >> 5;
  const int wave_m = wid & 3;
  const int wave_n = wid >> 2;
  const int m_blk  = blockIdx.x & 1;
  const int n_blk  = blockIdx.x >> 1;
  const int m_base = m_blk * 128;
  const int kh     = (lane >> 4) * 8;
  const int rA     = m_base + wave_m * 32 + (lane & 15);

  v8f acc[2][4];
  const v8f vz = {0.f, 0.f, 0.f, 0.f, 0.f, 0.f, 0.f, 0.f};
#pragma unroll
  for (int wm = 0; wm < 2; ++wm)
#pragma unroll
    for (int wn = 0; wn < 4; ++wn) acc[wm][wn] = vz;

  if constexpr (XM == 0) {
    // f32 input, K=32, convert in-register to bf16 fragments
    BF16Frag a[2], b[4];
#pragma unroll
    for (int wm = 0; wm < 2; ++wm) {
      const float* xr = xf + (size_t)(rA + wm * 16) * xstride;
      float4 f0 = *(const float4*)(xr + kh);
      float4 f1 = *(const float4*)(xr + kh + 4);
      float4 f2 = *(const float4*)(xr + 16 + kh);
      float4 f3 = *(const float4*)(xr + 16 + kh + 4);
      a[wm].u[0]  = f2bf(f0.x); a[wm].u[1]  = f2bf(f0.y); a[wm].u[2]  = f2bf(f0.z); a[wm].u[3]  = f2bf(f0.w);
      a[wm].u[4]  = f2bf(f1.x); a[wm].u[5]  = f2bf(f1.y); a[wm].u[6]  = f2bf(f1.z); a[wm].u[7]  = f2bf(f1.w);
      a[wm].u[8]  = f2bf(f2.x); a[wm].u[9]  = f2bf(f2.y); a[wm].u[10] = f2bf(f2.z); a[wm].u[11] = f2bf(f2.w);
      a[wm].u[12] = f2bf(f3.x); a[wm].u[13] = f2bf(f3.y); a[wm].u[14] = f2bf(f3.z); a[wm].u[15] = f2bf(f3.w);
    }
    load_bfrag(b, Wx + (size_t)n_blk * (8 * 32 * 16), 0, wave_n, lane);
#pragma unroll
    for (int wn = 0; wn < 4; ++wn) {
      acc[0][wn] = wmma_bf16(a[0].v, b[wn].v, acc[0][wn]);
      acc[1][wn] = wmma_bf16(a[1].v, b[wn].v, acc[1][wn]);
    }
  } else if constexpr (XM == 2) {
    // bf16 input, K=8 zero-padded to 32 (weights also zero-padded at pack time)
    BF16Frag a[2], b[4];
    uint4 z4 = make_uint4(0u, 0u, 0u, 0u);
    a[0].q[0] = z4; a[0].q[1] = z4; a[1].q[0] = z4; a[1].q[1] = z4;
    if (lane < 16) {  // only K=0..7 lanes carry data
      a[0].q[0] = *(const uint4*)(xh + (size_t)rA * OUTF);
      a[1].q[0] = *(const uint4*)(xh + (size_t)(rA + 16) * OUTF);
    }
    load_bfrag(b, Wx + (size_t)n_blk * (8 * 32 * 16), 0, wave_n, lane);
#pragma unroll
    for (int wn = 0; wn < 4; ++wn) {
      acc[0][wn] = wmma_bf16(a[0].v, b[wn].v, acc[0][wn]);
      acc[1][wn] = wmma_bf16(a[1].v, b[wn].v, acc[1][wn]);
    }
  } else {
    gemm_h512(acc, xh, Wx, rA, kh, wave_n, n_blk, lane);
  }
  gemm_h512(acc, hp, Wh, rA, kh, wave_n, n_blk, lane);
  lstm_epilogue(acc, bc, c, hout, Gs, m_base, n_blk, wave_m, wave_n, lane);
}

// decoder FC: out = h1 @ fcW.T + fcb ; also feeds bf16 autoregressive input
DEVINL void fc_step(const unsigned short* __restrict__ h1,
                    const float* __restrict__ fcW, const float* __restrict__ fcb,
                    float* __restrict__ out, unsigned short* __restrict__ xd, int s) {
  int tid  = threadIdx.x;
  int pair = tid >> 2;            // 64 (row, out-feature) pairs per block
  int part = tid & 3;             // 4-lane K split
  int b    = blockIdx.x * 8 + (pair >> 3);
  int of   = pair & 7;
  const unsigned short* hr = h1 + (size_t)b * HID + part * 128;
  const float* wr = fcW + (size_t)of * HID + part * 128;
  float sacc = 0.f;
#pragma unroll 8
  for (int k = 0; k < 128; ++k) sacc += bf2f(hr[k]) * wr[k];
  sacc += __shfl_xor(sacc, 1, 32);
  sacc += __shfl_xor(sacc, 2, 32);
  if (part == 0) {
    float v = sacc + fcb[of];
    out[((size_t)b * FUT + s) * OUTF + of] = v;
    xd[b * OUTF + of] = f2bf(v);
  }
}

// device-wide sense barrier (32 co-resident blocks)
DEVINL void grid_sync(unsigned* cnt, unsigned* gen, unsigned target) {
  __syncthreads();
  if (threadIdx.x == 0) {
    __threadfence();
    unsigned arrived = __hip_atomic_fetch_add(cnt, 1u, __ATOMIC_ACQ_REL, __HIP_MEMORY_SCOPE_AGENT);
    if (arrived == (unsigned)(NBLK - 1)) {
      __hip_atomic_store(cnt, 0u, __ATOMIC_RELAXED, __HIP_MEMORY_SCOPE_AGENT);
      __hip_atomic_fetch_add(gen, 1u, __ATOMIC_RELEASE, __HIP_MEMORY_SCOPE_AGENT);
    } else {
      while (__hip_atomic_load(gen, __ATOMIC_ACQUIRE, __HIP_MEMORY_SCOPE_AGENT) < target)
        __builtin_amdgcn_s_sleep(2);
    }
  }
  __syncthreads();
}

// ---------------- init: pack weights to fragment-native bf16, biases, states ----------------
__device__ void pack_w(unsigned short* dst, const float* src, int Ksrc, int KC,
                       size_t g, size_t n) {
  size_t tot = (size_t)GATES * KC * 32;  // GATES * Kpad elements
  for (size_t idx = g; idx < tot; idx += n) {
    int e = (int)(idx & 15);
    size_t t = idx >> 4;
    int l  = (int)(t & 31); t >>= 5;
    int ng = (int)(t & 7);  t >>= 3;
    int kc = (int)(t % KC);
    int nb = (int)(t / KC);
    int tc   = ng * 16 + (l & 15);
    int gcol = (tc >> 5) * HID + nb * 32 + (tc & 31);
    int K    = kc * 32 + (e < 8 ? (l >> 4) * 8 + e : 16 + (l >> 4) * 8 + (e - 8));
    float v  = (K < Ksrc) ? src[(size_t)gcol * Ksrc + K] : 0.f;
    dst[idx] = f2bf(v);
  }
}

__global__ void __launch_bounds__(THREADS) lstm_seq2seq_init(
    const float* __restrict__ in_seq,
    const float* eWih0, const float* eWhh0, const float* ebih0, const float* ebhh0,
    const float* eWih1, const float* eWhh1, const float* ebih1, const float* ebhh1,
    const float* dWih0, const float* dWhh0, const float* dbih0, const float* dbhh0,
    const float* dWih1, const float* dWhh1, const float* dbih1, const float* dbhh1,
    char* __restrict__ ws) {
  const size_t g = (size_t)blockIdx.x * blockDim.x + threadIdx.x;
  const size_t n = (size_t)gridDim.x * blockDim.x;

  pack_w((unsigned short*)(ws + OFF_PEWIH0), eWih0, 32,  1,  g, n);
  pack_w((unsigned short*)(ws + OFF_PEWHH0), eWhh0, 512, 16, g, n);
  pack_w((unsigned short*)(ws + OFF_PEWIH1), eWih1, 512, 16, g, n);
  pack_w((unsigned short*)(ws + OFF_PEWHH1), eWhh1, 512, 16, g, n);
  pack_w((unsigned short*)(ws + OFF_PDWIH0), dWih0, 8,   1,  g, n);
  pack_w((unsigned short*)(ws + OFF_PDWHH0), dWhh0, 512, 16, g, n);
  pack_w((unsigned short*)(ws + OFF_PDWIH1), dWih1, 512, 16, g, n);
  pack_w((unsigned short*)(ws + OFF_PDWHH1), dWih1 == dWih1 ? dWhh1 : dWhh1, 512, 16, g, n);

  float* bc0 = (float*)(ws + OFF_BC0);
  float* bc1 = (float*)(ws + OFF_BC1);
  float* bc2 = (float*)(ws + OFF_BC2);
  float* bc3 = (float*)(ws + OFF_BC3);
  for (size_t i = g; i < (size_t)GATES; i += n) {
    bc0[i] = ebih0[i] + ebhh0[i];
    bc1[i] = ebih1[i] + ebhh1[i];
    bc2[i] = dbih0[i] + dbhh0[i];
    bc3[i] = dbih1[i] + dbhh1[i];
  }

  unsigned short* hz = (unsigned short*)(ws + OFF_H0A);  // H0A..H1B contiguous
  for (size_t i = g; i < 4 * (size_t)BATCH * HID; i += n) hz[i] = 0;
  float* cz = (float*)(ws + OFF_C0);                     // C0,C1 contiguous
  for (size_t i = g; i < 2 * (size_t)BATCH * HID; i += n) cz[i] = 0.f;

  unsigned short* xd = (unsigned short*)(ws + OFF_XD);   // dec_in = input_seq[:, -1, :8]
  for (size_t i = g; i < (size_t)BATCH * OUTF; i += n) {
    size_t b = i >> 3;
    int j = (int)(i & 7);
    xd[i] = f2bf(in_seq[b * (size_t)(TSEQ * INF) + (size_t)(TSEQ - 1) * INF + j]);
  }
}

// ---------------- persistent main kernel ----------------
__global__ void __launch_bounds__(THREADS) lstm_seq2seq_main(
    const float* __restrict__ in_seq, const float* __restrict__ fcW,
    const float* __restrict__ fcb, char* __restrict__ ws, float* __restrict__ out) {
  __shared__ float Gs[128 * 132];

  unsigned* cnt = (unsigned*)(ws + OFF_SYNC);
  unsigned* gen = cnt + 1;
  unsigned bar = 0;

  float* C0 = (float*)(ws + OFF_C0);
  float* C1 = (float*)(ws + OFF_C1);
  unsigned short* XD = (unsigned short*)(ws + OFF_XD);
  const float* BC0 = (const float*)(ws + OFF_BC0);
  const float* BC1 = (const float*)(ws + OFF_BC1);
  const float* BC2 = (const float*)(ws + OFF_BC2);
  const float* BC3 = (const float*)(ws + OFF_BC3);
  const unsigned short* PEWIH0 = (const unsigned short*)(ws + OFF_PEWIH0);
  const unsigned short* PEWHH0 = (const unsigned short*)(ws + OFF_PEWHH0);
  const unsigned short* PEWIH1 = (const unsigned short*)(ws + OFF_PEWIH1);
  const unsigned short* PEWHH1 = (const unsigned short*)(ws + OFF_PEWHH1);
  const unsigned short* PDWIH0 = (const unsigned short*)(ws + OFF_PDWIH0);
  const unsigned short* PDWHH0 = (const unsigned short*)(ws + OFF_PDWHH0);
  const unsigned short* PDWIH1 = (const unsigned short*)(ws + OFF_PDWIH1);
  const unsigned short* PDWHH1 = (const unsigned short*)(ws + OFF_PDWHH1);

  size_t p0 = 0, p1 = 0;  // ping-pong as BYTE OFFSETS (keeps loads in global addrspace)

  // ---- encoder: 2 stacked LSTM layers over T timesteps ----
  for (int t = 0; t < TSEQ; ++t) {
    {
      const unsigned short* h0r = (const unsigned short*)(ws + OFF_H0A + p0);
      unsigned short*       h0w = (unsigned short*)(ws + OFF_H0A + (p0 ^ SZ_H));
      lstm_step<0>(in_seq + (size_t)t * INF, TSEQ * INF, nullptr,
                   PEWIH0, PEWHH0, BC0, h0r, C0, h0w, Gs);
      p0 ^= SZ_H;
    }
    grid_sync(cnt, gen, ++bar);
    {
      const unsigned short* x1  = (const unsigned short*)(ws + OFF_H0A + p0);
      const unsigned short* h1r = (const unsigned short*)(ws + OFF_H1A + p1);
      unsigned short*       h1w = (unsigned short*)(ws + OFF_H1A + (p1 ^ SZ_H));
      lstm_step<1>(nullptr, 0, x1, PEWIH1, PEWHH1, BC1, h1r, C1, h1w, Gs);
      p1 ^= SZ_H;
    }
    grid_sync(cnt, gen, ++bar);
  }

  // ---- autoregressive decoder: 2 stacked LSTM layers + FC, FUT steps ----
  for (int s = 0; s < FUT; ++s) {
    {
      const unsigned short* h0r = (const unsigned short*)(ws + OFF_H0A + p0);
      unsigned short*       h0w = (unsigned short*)(ws + OFF_H0A + (p0 ^ SZ_H));
      lstm_step<2>(nullptr, 0, XD, PDWIH0, PDWHH0, BC2, h0r, C0, h0w, Gs);
      p0 ^= SZ_H;
    }
    grid_sync(cnt, gen, ++bar);
    {
      const unsigned short* x1  = (const unsigned short*)(ws + OFF_H0A + p0);
      const unsigned short* h1r = (const unsigned short*)(ws + OFF_H1A + p1);
      unsigned short*       h1w = (unsigned short*)(ws + OFF_H1A + (p1 ^ SZ_H));
      lstm_step<1>(nullptr, 0, x1, PDWIH1, PDWHH1, BC3, h1r, C1, h1w, Gs);
      p1 ^= SZ_H;
    }
    grid_sync(cnt, gen, ++bar);
    fc_step((const unsigned short*)(ws + OFF_H1A + p1), fcW, fcb, out, XD, s);
    grid_sync(cnt, gen, ++bar);
  }
}

// ---------------- launch ----------------
extern "C" void kernel_launch(void* const* d_in, const int* in_sizes, int n_in,
                              void* d_out, int out_size, void* d_ws, size_t ws_size,
                              hipStream_t stream) {
  (void)in_sizes; (void)n_in; (void)out_size; (void)ws_size;
  const float* in_seq = (const float*)d_in[0];
  const float* eWih0  = (const float*)d_in[1];
  const float* eWhh0  = (const float*)d_in[2];
  const float* ebih0  = (const float*)d_in[3];
  const float* ebhh0  = (const float*)d_in[4];
  const float* eWih1  = (const float*)d_in[5];
  const float* eWhh1  = (const float*)d_in[6];
  const float* ebih1  = (const float*)d_in[7];
  const float* ebhh1  = (const float*)d_in[8];
  const float* dWih0  = (const float*)d_in[9];
  const float* dWhh0  = (const float*)d_in[10];
  const float* dbih0  = (const float*)d_in[11];
  const float* dbhh0  = (const float*)d_in[12];
  const float* dWih1  = (const float*)d_in[13];
  const float* dWhh1  = (const float*)d_in[14];
  const float* dbih1  = (const float*)d_in[15];
  const float* dbhh1  = (const float*)d_in[16];
  const float* fcW    = (const float*)d_in[17];
  const float* fcb    = (const float*)d_in[18];
  char* ws = (char*)d_ws;

  hipMemsetAsync(ws + OFF_SYNC, 0, 256, stream);  // reset grid barrier
  lstm_seq2seq_init<<<dim3(256), dim3(THREADS), 0, stream>>>(
      in_seq, eWih0, eWhh0, ebih0, ebhh0, eWih1, eWhh1, ebih1, ebhh1,
      dWih0, dWhh0, dbih0, dbhh0, dWih1, dWhh1, dbih1, dbhh1, ws);
  lstm_seq2seq_main<<<dim3(NBLK), dim3(THREADS), 0, stream>>>(
      in_seq, fcW, fcb, ws, (float*)d_out);
}
